// GraphTransformer_17635135718119
// MI455X (gfx1250) — matrix operand order repaired and checked by reading, not compiled, not run on hardware
//
#include <hip/hip_runtime.h>

// ---------------------------------------------------------------------------
// Problem constants (match reference)
// ---------------------------------------------------------------------------
#define NN   50000
#define EE   800000
#define NH   4

typedef __attribute__((ext_vector_type(16))) __bf16 bf16x16;
typedef __attribute__((ext_vector_type(8)))  float  floatx8;

// ---------------------------------------------------------------------------
// Weight pre-pack: W [128, ncols] fp32 row-major -> fragment-ordered bf16.
//   Packed index t = ((nt*4 + kk)*32 + lane)*16 + j
//   maps to W[k][n] with n = nt*16 + (lane&15), k = kk*32 + (lane>>4)*16 + j,
//   which is the contiguous k order (k = 16h + j) this code also uses for the A fragment.
// ---------------------------------------------------------------------------
__global__ void __launch_bounds__(256)
pack_w(const float* __restrict__ W, __bf16* __restrict__ Wp, int ncols, int total)
{
    int t = blockIdx.x * blockDim.x + threadIdx.x;
    if (t >= total) return;
    const int j    =  t        & 15;
    const int lane = (t >> 4)  & 31;
    const int kk   = (t >> 9)  & 3;
    const int nt   =  t >> 11;
    const int n = nt * 16 + (lane & 15);
    const int k = kk * 32 + (lane >> 4) * 16 + j;
    Wp[t] = (__bf16)W[k * ncols + n];
}

// ---------------------------------------------------------------------------
// WMMA bf16 GEMM:  out[m, ocol0+n] = sum_k A[m,k] * W[k,n] + bias[n]
//   A: [M,128] fp32. Wp: fragment-packed bf16 (see pack_w). K = 128 fixed.
//   Block = 8 waves; A tile (16x128) staged once in LDS in fragment order,
//   each wave computes one 16x16 tile with 4 x v_wmma_f32_16x16x32_bf16.
//   B fragment = 1 contiguous 32B load; A fragment = 1 contiguous 32B LDS read.
// ---------------------------------------------------------------------------
__global__ void __launch_bounds__(256)
gemm_bf16_wmma(const float* __restrict__ A, const __bf16* __restrict__ Wp,
               const float* __restrict__ bias, float* __restrict__ out,
               int ncols, int ostride, int ocol0)
{
    __shared__ __align__(32) __bf16 aFrag[4][32][16];   // 4 KB, fragment order

    const int t    = threadIdx.x;
    const int lane = t & 31;
    const int wave = t >> 5;
    const int m0   = blockIdx.x * 16;

    // ---- cooperative A-tile stage: 256 threads x 8 floats -> bf16 LDS ------
    {
        const int kk   =  t >> 6;         // 0..3   (k0 = kk*32)
        const int ln   = (t >> 1) & 31;   // fragment lane
        const int half =  t & 1;          // low/high 8 elements of fragment
        const int m    = ln & 15;
        const int hi   = ln >> 4;
        // contiguous k order used here (as pack_w): lanes 0-15 -> K = k0+{0..15},
        //                        lanes16-31 -> K = k0+{16..31}
        const int kb = kk * 32 + hi * 16 + half * 8;
        const float* ar = A + (size_t)(m0 + m) * 128 + kb;
        const float4 f0 = *(const float4*)(ar);
        const float4 f1 = *(const float4*)(ar + 4);
        __bf16* dp = &aFrag[kk][ln][half * 8];
        dp[0] = (__bf16)f0.x; dp[1] = (__bf16)f0.y;
        dp[2] = (__bf16)f0.z; dp[3] = (__bf16)f0.w;
        dp[4] = (__bf16)f1.x; dp[5] = (__bf16)f1.y;
        dp[6] = (__bf16)f1.z; dp[7] = (__bf16)f1.w;
    }
    __syncthreads();

    const int ntile = blockIdx.y * 8 + wave;
    if (ntile * 16 < ncols) {                 // wave-uniform: EXEC stays all-1s
        const int l15  = lane & 15;
        const int hi   = lane >> 4;
        const int ncol = ntile * 16 + l15;

        floatx8 acc = {};
        const __bf16* wp = Wp + ((size_t)ntile * 4) * 512 + lane * 16;
#pragma unroll
        for (int kk = 0; kk < 4; ++kk) {
            bf16x16 a = *(const bf16x16*)&aFrag[kk][lane][0];
            bf16x16 b = *(const bf16x16*)(wp + kk * 512);
            acc = __builtin_amdgcn_wmma_f32_16x16x32_bf16(
                      false, a, false, b, (short)0, acc, false, false);
        }

        const float bn = bias[ncol];
#pragma unroll
        for (int r = 0; r < 8; ++r) {
            const int m = m0 + r + hi * 8;    // C/D: VGPR r -> M = r + hi*8
            out[(size_t)m * ostride + ocol0 + ncol] = acc[r] + bn;
        }
    }
}

// ---------------------------------------------------------------------------
// Monotone float<->uint mapping for atomicMax on floats (handles negatives)
// ---------------------------------------------------------------------------
__device__ __forceinline__ unsigned fkey(float f)
{
    unsigned u = __float_as_uint(f);
    return (u & 0x80000000u) ? ~u : (u | 0x80000000u);
}
__device__ __forceinline__ float funkey(unsigned k)
{
    unsigned u = (k & 0x80000000u) ? (k & 0x7fffffffu) : ~k;
    return __uint_as_float(u);
}
#define NEG_INF_KEY 0x007FFFFFu   // fkey(-inf)

// ---------------------------------------------------------------------------
// Init per-layer accumulators: nmax <- key(-inf), nsum <- 0, agg <- 0
// ---------------------------------------------------------------------------
__global__ void init_accum(unsigned* __restrict__ nmax, float* __restrict__ nsum,
                           float* __restrict__ agg, int nh, int aggCount)
{
    int t = blockIdx.x * blockDim.x + threadIdx.x;
    if (t < nh) { nmax[t] = NEG_INF_KEY; nsum[t] = 0.0f; }
    if (t < aggCount) agg[t] = 0.0f;
}

// ---------------------------------------------------------------------------
// Edge pass 1: alpha[e,h] = (q[dst]·k[src]) * scale ; running max per (dst,h)
// ---------------------------------------------------------------------------
__global__ void __launch_bounds__(256)
edge_alpha(const float* __restrict__ qkvs, const long long* __restrict__ srcv,
           const long long* __restrict__ dstv, float* __restrict__ alpha,
           unsigned* __restrict__ nmax, int C, int stride, int qoff, int koff,
           float scale)
{
    int t = blockIdx.x * blockDim.x + threadIdx.x;
    if (t >= EE * NH) return;
    const int e = t >> 2, h = t & 3;
    const int s = (int)srcv[e], d = (int)dstv[e];       // 32-bit index math
    const float* q = qkvs + d * stride + qoff + h * C;
    const float* k = qkvs + s * stride + koff + h * C;
    float acc = 0.0f;
    for (int c = 0; c < C; c += 4) {
        float4 qa = *(const float4*)(q + c);
        float4 ka = *(const float4*)(k + c);
        acc += qa.x * ka.x + qa.y * ka.y + qa.z * ka.z + qa.w * ka.w;
    }
    acc *= scale;
    alpha[t] = acc;
    atomicMax(nmax + d * NH + h, fkey(acc));
}

// ---------------------------------------------------------------------------
// Edge pass 2: ex = exp(alpha - max[dst,h]) (in place); running sum per (dst,h)
// ---------------------------------------------------------------------------
__global__ void __launch_bounds__(256)
edge_exp(float* __restrict__ alpha, const long long* __restrict__ dstv,
         const unsigned* __restrict__ nmax, float* __restrict__ nsum)
{
    int t = blockIdx.x * blockDim.x + threadIdx.x;
    if (t >= EE * NH) return;
    const int e = t >> 2, h = t & 3;
    const int d = (int)dstv[e];
    const float m  = funkey(nmax[d * NH + h]);
    const float ex = __expf(alpha[t] - m);
    alpha[t] = ex;
    atomicAdd(nsum + d * NH + h, ex);
}

// ---------------------------------------------------------------------------
// Edge pass 3: agg[dst,h,:] += (ex / sum[dst,h]) * v[src,h,:]
// ---------------------------------------------------------------------------
__global__ void __launch_bounds__(256)
edge_scatter(const float* __restrict__ alpha, const float* __restrict__ qkvs,
             const long long* __restrict__ srcv, const long long* __restrict__ dstv,
             const float* __restrict__ nsum, float* __restrict__ agg,
             int C, int stride, int voff, int aggStride)
{
    int t = blockIdx.x * blockDim.x + threadIdx.x;
    if (t >= EE * NH) return;
    const int e = t >> 2, h = t & 3;
    const int s = (int)srcv[e], d = (int)dstv[e];
    const float w = alpha[t] / (nsum[d * NH + h] + 1e-16f);
    const float* v = qkvs + s * stride + voff + h * C;
    float*       o = agg  + d * aggStride + h * C;
    for (int c = 0; c < C; c += 4) {
        float4 va = *(const float4*)(v + c);
        atomicAdd(o + c + 0, w * va.x);
        atomicAdd(o + c + 1, w * va.y);
        atomicAdd(o + c + 2, w * va.z);
        atomicAdd(o + c + 3, w * va.w);
    }
}

// ---------------------------------------------------------------------------
// Node combine for layers 0/1 (concat=True): beta-gate + LayerNorm + ReLU.
// One 128-thread block per node, LDS tree reductions.
// ---------------------------------------------------------------------------
__global__ void __launch_bounds__(128)
node_combine_ln(const float* __restrict__ agg, const float* __restrict__ qkvs,
                const float* __restrict__ Wbeta, const float* __restrict__ g,
                const float* __restrict__ bln, float* __restrict__ outh,
                int stride, int skipoff)
{
    const int n = blockIdx.x, t = threadIdx.x;
    __shared__ float red[128];

    const float o  = agg[n * 128 + t];
    const float xr = qkvs[(size_t)n * stride + skipoff + t];

    // beta = sigmoid([out, x_r, out - x_r] @ Wbeta)
    red[t] = o * Wbeta[t] + xr * Wbeta[128 + t] + (o - xr) * Wbeta[256 + t];
    __syncthreads();
    for (int s = 64; s > 0; s >>= 1) { if (t < s) red[t] += red[t + s]; __syncthreads(); }
    const float beta = 1.0f / (1.0f + __expf(-red[0]));
    __syncthreads();

    const float hv = beta * xr + (1.0f - beta) * o;

    // LayerNorm over 128 features
    red[t] = hv; __syncthreads();
    for (int s = 64; s > 0; s >>= 1) { if (t < s) red[t] += red[t + s]; __syncthreads(); }
    const float mu = red[0] * (1.0f / 128.0f);
    __syncthreads();
    const float dv = hv - mu;
    red[t] = dv * dv; __syncthreads();
    for (int s = 64; s > 0; s >>= 1) { if (t < s) red[t] += red[t + s]; __syncthreads(); }
    const float var = red[0] * (1.0f / 128.0f);

    const float y = dv * rsqrtf(var + 1e-5f) * g[t] + bln[t];
    outh[n * 128 + t] = fmaxf(y, 0.0f);                  // ReLU
}

// ---------------------------------------------------------------------------
// Final node combine (layer 2, concat=False): head-mean + beta-gate -> d_out
// ---------------------------------------------------------------------------
__global__ void __launch_bounds__(64)
node_combine_final(const float* __restrict__ agg, const float* __restrict__ qkvs,
                   const float* __restrict__ Wbeta2, float* __restrict__ out)
{
    const int n = blockIdx.x, t = threadIdx.x;           // t = 0..63
    __shared__ float red[64];
    const int base = n * 256;
    const float o = 0.25f * (agg[base + t] + agg[base + 64 + t] +
                             agg[base + 128 + t] + agg[base + 192 + t]);
    const float xr = qkvs[(size_t)n * 832 + 768 + t];

    red[t] = o * Wbeta2[t] + xr * Wbeta2[64 + t] + (o - xr) * Wbeta2[128 + t];
    __syncthreads();
    for (int s = 32; s > 0; s >>= 1) { if (t < s) red[t] += red[t + s]; __syncthreads(); }
    const float beta = 1.0f / (1.0f + __expf(-red[0]));
    out[n * 64 + t] = beta * xr + (1.0f - beta) * o;
}

// ---------------------------------------------------------------------------
// Host-side orchestration
// ---------------------------------------------------------------------------
extern "C" void kernel_launch(void* const* d_in, const int* in_sizes, int n_in,
                              void* d_out, int out_size, void* d_ws, size_t ws_size,
                              hipStream_t stream)
{
    (void)in_sizes; (void)n_in; (void)out_size; (void)ws_size;

    // ---- inputs (setup_inputs order) ----
    const float*     x       = (const float*)d_in[0];
    const long long* ei      = (const long long*)d_in[1];   // int64 [2,E]
    const float*     Wq01    = (const float*)d_in[2];
    const float*     bq01    = (const float*)d_in[3];
    const float*     Wk01    = (const float*)d_in[4];
    const float*     bk01    = (const float*)d_in[5];
    const float*     Wv01    = (const float*)d_in[6];
    const float*     bv01    = (const float*)d_in[7];
    const float*     Wskip01 = (const float*)d_in[8];
    const float*     bskip01 = (const float*)d_in[9];
    const float*     Wbeta01 = (const float*)d_in[10];
    const float*     ln_g    = (const float*)d_in[11];
    const float*     ln_b    = (const float*)d_in[12];
    const float*     Wq2     = (const float*)d_in[13];
    const float*     bq2     = (const float*)d_in[14];
    const float*     Wk2     = (const float*)d_in[15];
    const float*     bk2     = (const float*)d_in[16];
    const float*     Wv2     = (const float*)d_in[17];
    const float*     bv2     = (const float*)d_in[18];
    const float*     Wskip2  = (const float*)d_in[19];
    const float*     bskip2  = (const float*)d_in[20];
    const float*     Wbeta2  = (const float*)d_in[21];

    const long long* srcv = ei;
    const long long* dstv = ei + EE;

    // ---- workspace carve-out (256 B aligned) ----
    char*  base = (char*)d_ws;
    size_t off  = 0;
    auto carve = [&](size_t bytes) -> char* {
        char* p = base + off;
        off = (off + bytes + 255) & ~(size_t)255;
        return p;
    };
    float*    qkvs  = (float*)   carve((size_t)NN * 832 * 4);  // fused q|k|v|skip
    float*    agg   = (float*)   carve((size_t)NN * 256 * 4);  // attention output
    float*    alpha = (float*)   carve((size_t)EE * NH * 4);   // alpha -> ex in place
    unsigned* nmax  = (unsigned*)carve((size_t)NN * NH * 4);
    float*    nsum  = (float*)   carve((size_t)NN * NH * 4);
    float*    hbuf  = (float*)   carve((size_t)NN * 128 * 4);  // hidden state
    // packed bf16 weights (fragment order)
    __bf16* Wp01[8];                                           // q,k,v,skip x 2 layers
    for (int i = 0; i < 8; ++i) Wp01[i] = (__bf16*)carve(128 * 128 * 2);
    __bf16* Wpq2 = (__bf16*)carve(128 * 256 * 2);
    __bf16* Wpk2 = (__bf16*)carve(128 * 256 * 2);
    __bf16* Wpv2 = (__bf16*)carve(128 * 256 * 2);
    __bf16* Wps2 = (__bf16*)carve(128 * 64 * 2);

    const dim3 blk(256);
    const int  Mtiles = NN / 16;                               // 3125 exactly
    const dim3 edgeGrid((EE * NH + 255) / 256);
    const float scale01 = 0.17677669529663688f;                // 1/sqrt(32)
    const float scale2  = 0.125f;                              // 1/sqrt(64)

    // ---- one-shot weight packing (graph-capture safe, deterministic) -------
    {
        const float* Wsrc01[8] = { Wq01, Wq01 + 16384, Wk01, Wk01 + 16384,
                                   Wv01, Wv01 + 16384, Wskip01, Wskip01 + 16384 };
        // order: [q L0, q L1, k L0, k L1, v L0, v L1, s L0, s L1] -> remap below
        for (int i = 0; i < 8; ++i)
            pack_w<<<dim3((128 * 128 + 255) / 256), blk, 0, stream>>>(
                Wsrc01[i], Wp01[i], 128, 128 * 128);
        pack_w<<<dim3((128 * 256 + 255) / 256), blk, 0, stream>>>(Wq2, Wpq2, 256, 128 * 256);
        pack_w<<<dim3((128 * 256 + 255) / 256), blk, 0, stream>>>(Wk2, Wpk2, 256, 128 * 256);
        pack_w<<<dim3((128 * 256 + 255) / 256), blk, 0, stream>>>(Wv2, Wpv2, 256, 128 * 256);
        pack_w<<<dim3((128 * 64  + 255) / 256), blk, 0, stream>>>(Wskip2, Wps2, 64, 128 * 64);
    }

    // ================= layers 0 and 1 (HID=128, concat=True) =================
    for (int i = 0; i < 2; ++i) {
        const float* A = (i == 0) ? x : hbuf;

        init_accum<<<dim3((NN * 128 + 255) / 256), blk, 0, stream>>>(
            nmax, nsum, agg, NN * NH, NN * 128);

        const __bf16* Wps[4] = { Wp01[0 + i], Wp01[2 + i], Wp01[4 + i], Wp01[6 + i] };
        const float*  bs[4]  = { bq01 + i * 128, bk01 + i * 128,
                                 bv01 + i * 128, bskip01 + i * 128 };
        for (int p = 0; p < 4; ++p)
            gemm_bf16_wmma<<<dim3(Mtiles, 1), blk, 0, stream>>>(
                A, Wps[p], bs[p], qkvs, 128, 512, p * 128);

        edge_alpha<<<edgeGrid, blk, 0, stream>>>(qkvs, srcv, dstv, alpha, nmax,
                                                 32, 512, 0, 128, scale01);
        edge_exp<<<edgeGrid, blk, 0, stream>>>(alpha, dstv, nmax, nsum);
        edge_scatter<<<edgeGrid, blk, 0, stream>>>(alpha, qkvs, srcv, dstv, nsum,
                                                   agg, 32, 512, 256, 128);

        node_combine_ln<<<dim3(NN), dim3(128), 0, stream>>>(
            agg, qkvs, Wbeta01 + (size_t)i * 384, ln_g + i * 128, ln_b + i * 128,
            hbuf, 512, 384);
    }

    // ================= layer 2 (per-head OUT=64, concat=False) ================
    init_accum<<<dim3((NN * 256 + 255) / 256), blk, 0, stream>>>(
        nmax, nsum, agg, NN * NH, NN * 256);

    gemm_bf16_wmma<<<dim3(Mtiles, 2), blk, 0, stream>>>(hbuf, Wpq2, bq2, qkvs, 256, 832, 0);
    gemm_bf16_wmma<<<dim3(Mtiles, 2), blk, 0, stream>>>(hbuf, Wpk2, bk2, qkvs, 256, 832, 256);
    gemm_bf16_wmma<<<dim3(Mtiles, 2), blk, 0, stream>>>(hbuf, Wpv2, bv2, qkvs, 256, 832, 512);
    gemm_bf16_wmma<<<dim3(Mtiles, 1), blk, 0, stream>>>(hbuf, Wps2, bskip2, qkvs, 64, 832, 768);

    edge_alpha<<<edgeGrid, blk, 0, stream>>>(qkvs, srcv, dstv, alpha, nmax,
                                             64, 832, 0, 256, scale2);
    edge_exp<<<edgeGrid, blk, 0, stream>>>(alpha, dstv, nmax, nsum);
    edge_scatter<<<edgeGrid, blk, 0, stream>>>(alpha, qkvs, srcv, dstv, nsum,
                                               agg, 64, 832, 512, 256);

    node_combine_final<<<dim3(NN), dim3(64), 0, stream>>>(agg, qkvs, Wbeta2,
                                                          (float*)d_out);
}
